// LBONorm_19464791786011
// MI455X (gfx1250) — matrix operand order repaired and checked by reading, not compiled, not run on hardware
//
#include <hip/hip_runtime.h>
#include <hip/hip_bf16.h>
#include <math.h>

#define DIM  1024
#define RANK 128
#define ROWS 16          // rows per block
#define YPAD 1032        // 1024 + 8 halfs -> row stride 516 dwords (mod 64 == 4): conflict-free A loads
#define TPAD 136         // 128 + 8 halfs  -> row stride  68 dwords (mod 64 == 4)

typedef __attribute__((ext_vector_type(16))) _Float16 v16h;
typedef __attribute__((ext_vector_type(8)))  float    v8f;

union AFrag { v16h h; unsigned int u[8]; };

// ---------------------------------------------------------------------------
// Kernel 1: power iteration -> h_val (single block, tiny)
// ---------------------------------------------------------------------------
__global__ __launch_bounds__(256) void lbo_hval_kernel(
    const float* __restrict__ V, const float* __restrict__ sv,
    const float* __restrict__ hptr, float* __restrict__ hval_out)
{
    __shared__ float u[RANK];
    __shared__ float vnew[DIM];
    __shared__ float red[256];
    const int tid = threadIdx.x;

    // u = V @ sv
    if (tid < RANK) {
        float acc = 0.f;
        const float* row = V + tid * DIM;
        for (int d = 0; d < DIM; ++d) acc += row[d] * sv[d];
        u[tid] = acc;
    }
    __syncthreads();

    // ||u||, normalize
    red[tid] = (tid < RANK) ? u[tid] * u[tid] : 0.f;
    __syncthreads();
    for (int s = 128; s > 0; s >>= 1) { if (tid < s) red[tid] += red[tid + s]; __syncthreads(); }
    float un = fmaxf(sqrtf(red[0]), 1e-12f);
    __syncthreads();
    if (tid < RANK) u[tid] /= un;
    __syncthreads();

    // vnew = V^T @ u, accumulate ||vnew||^2
    float nrm = 0.f;
    for (int d = tid; d < DIM; d += 256) {
        float acc = 0.f;
        for (int r = 0; r < RANK; ++r) acc += V[r * DIM + d] * u[r];
        vnew[d] = acc;
        nrm += acc * acc;
    }
    red[tid] = nrm;
    __syncthreads();
    for (int s = 128; s > 0; s >>= 1) { if (tid < s) red[tid] += red[tid + s]; __syncthreads(); }
    float vn = fmaxf(sqrtf(red[0]), 1e-12f);
    __syncthreads();
    for (int d = tid; d < DIM; d += 256) vnew[d] /= vn;
    __syncthreads();

    // sigma^2 = || V @ vnew ||^2
    float w2 = 0.f;
    if (tid < RANK) {
        float acc = 0.f;
        const float* row = V + tid * DIM;
        for (int d = 0; d < DIM; ++d) acc += row[d] * vnew[d];
        w2 = acc * acc;
    }
    red[tid] = w2;
    __syncthreads();
    for (int s = 128; s > 0; s >>= 1) { if (tid < s) red[tid] += red[tid + s]; __syncthreads(); }
    if (tid == 0) {
        float sigma2 = red[0];
        float hmax = 1.f / (sigma2 + 1e-6f);
        hval_out[0] = fminf(fabsf(hptr[0]), hmax);
    }
}

// ---------------------------------------------------------------------------
// Kernel 2: V f32 -> f16 row-major (Vh[r][d]) and f16 transposed (Vt[d][r])
// ---------------------------------------------------------------------------
__global__ __launch_bounds__(256) void lbo_cvt_kernel(
    const float* __restrict__ V, _Float16* __restrict__ Vh, _Float16* __restrict__ Vt)
{
    int i = blockIdx.x * 256 + threadIdx.x;   // i in [0, RANK*DIM)
    int r = i >> 10;
    int d = i & (DIM - 1);
    _Float16 hv = (_Float16)V[i];
    Vh[i] = hv;
    Vt[d * RANK + r] = hv;
}

// ---------------------------------------------------------------------------
// Kernel 3: fused LN + low-rank projection (WMMA f32<-f16) + epilogue
// ---------------------------------------------------------------------------
__global__ __launch_bounds__(256) void lbo_main_kernel(
    const float* __restrict__ x, const float* __restrict__ scale,
    const float* __restrict__ bias, const float* __restrict__ alpha_ptr,
    const float* __restrict__ hval_ptr,
    const _Float16* __restrict__ Vh, const _Float16* __restrict__ Vt,
    float* __restrict__ out)
{
    __shared__ _Float16 y16[ROWS][YPAD];
    __shared__ _Float16 T16[ROWS][TPAD];
    __shared__ float muS[ROWS], riS[ROWS], confS[ROWS];

    const int tid  = threadIdx.x;
    const int wave = tid >> 5;        // 0..7
    const int lane = tid & 31;
    const int l16  = lane & 15;
    const int hi   = (lane >> 4) & 1; // 0 for lanes 0-15, 1 for lanes 16-31
    const int row0 = blockIdx.x * ROWS;

    const float alpha = fabsf(alpha_ptr[0]);
    const float hv    = hval_ptr[0];

    // ---- Phase 1: LayerNorm stats, write y (f16) to LDS; 2 rows per wave ----
    for (int rr = 0; rr < 2; ++rr) {
        const int s = wave * 2 + rr;
        const float* xr = x + (size_t)(row0 + s) * DIM;
        float xs[32];
        float sum = 0.f;
        #pragma unroll
        for (int j = 0; j < 32; ++j) { xs[j] = xr[j * 32 + lane]; sum += xs[j]; }
        #pragma unroll
        for (int o = 16; o > 0; o >>= 1) sum += __shfl_xor(sum, o, 32);
        const float mu = sum * (1.f / 1024.f);
        float ss = 0.f;
        #pragma unroll
        for (int j = 0; j < 32; ++j) { float t = xs[j] - mu; ss += t * t; }
        #pragma unroll
        for (int o = 16; o > 0; o >>= 1) ss += __shfl_xor(ss, o, 32);
        const float var = ss * (1.f / 1024.f);
        const float ri  = rsqrtf(var + 1e-5f);
        #pragma unroll
        for (int j = 0; j < 32; ++j)
            y16[s][j * 32 + lane] = (_Float16)((xs[j] - mu) * ri);
        if (lane == 0) {
            muS[s] = mu;
            riS[s] = ri;
            confS[s] = expf(-2.f * alpha * (ss * ri * ri));  // sum(y^2) = ss/(var+eps)
        }
    }
    __syncthreads();

    // ---- Phase 2: T[16 x 128] = y @ V^T ; wave w owns rank tile [16w,16w+16) ----
    {
        v8f acc = {0.f, 0.f, 0.f, 0.f, 0.f, 0.f, 0.f, 0.f};
        const int rrow = wave * 16 + l16;               // B operand: V row r
        for (int k0 = 0; k0 < DIM; k0 += 32) {
            AFrag a;
            #pragma unroll
            for (int v = 0; v < 8; ++v) {
                int k = k0 + hi * 16 + (v << 1);
                a.u[v] = *(const unsigned int*)&y16[l16][k];
            }
            v16h b = *(const v16h*)(Vh + (size_t)rrow * DIM + k0 + hi * 16);
            acc = __builtin_amdgcn_wmma_f32_16x16x32_f16(
                      false, a.h, false, b, (short)0, acc, false, false);
        }
        // D layout: VGPR j -> M = j + 8*hi, N = l16  ==> T[s = M][r = 16*wave + N]
        #pragma unroll
        for (int j = 0; j < 8; ++j)
            T16[j + hi * 8][wave * 16 + l16] = (_Float16)acc[j];
    }
    __syncthreads();

    // ---- Phase 3: xW tile = T @ V ; epilogue; wave w owns 8 d-tiles ----
    for (int t = 0; t < 8; ++t) {
        const int dbase = (wave * 8 + t) * 16;
        v8f acc = {0.f, 0.f, 0.f, 0.f, 0.f, 0.f, 0.f, 0.f};
        #pragma unroll
        for (int k0 = 0; k0 < RANK; k0 += 32) {
            AFrag a;
            #pragma unroll
            for (int v = 0; v < 8; ++v) {
                int k = k0 + hi * 16 + (v << 1);
                a.u[v] = *(const unsigned int*)&T16[l16][k];
            }
            v16h b = *(const v16h*)(Vt + (size_t)(dbase + l16) * RANK + k0 + hi * 16);
            acc = __builtin_amdgcn_wmma_f32_16x16x32_f16(
                      false, a.h, false, b, (short)0, acc, false, false);
        }
        const int d  = dbase + l16;
        const float sc = scale[d];
        const float bi = bias[d];
        #pragma unroll
        for (int j = 0; j < 8; ++j) {
            const int s = j + hi * 8;
            const size_t gidx = (size_t)(row0 + s) * DIM + d;
            const float y = (x[gidx] - muS[s]) * riS[s];     // exact f32 y
            const float o = ((1.f - hv) * y + hv * confS[s] * acc[j]) * sc + bi;
            __builtin_nontemporal_store(o, &out[gidx]);
        }
    }
}

// ---------------------------------------------------------------------------
extern "C" void kernel_launch(void* const* d_in, const int* in_sizes, int n_in,
                              void* d_out, int out_size, void* d_ws, size_t ws_size,
                              hipStream_t stream) {
    const float* x     = (const float*)d_in[0];
    const float* V     = (const float*)d_in[1];
    const float* h     = (const float*)d_in[2];
    const float* scale = (const float*)d_in[3];
    const float* bias  = (const float*)d_in[4];
    const float* alpha = (const float*)d_in[5];
    const float* sv    = (const float*)d_in[6];
    float* out = (float*)d_out;

    char* ws = (char*)d_ws;
    float*    hval = (float*)ws;
    _Float16* Vh   = (_Float16*)(ws + 256);
    _Float16* Vt   = (_Float16*)(ws + 256 + (size_t)RANK * DIM * sizeof(_Float16));

    lbo_hval_kernel<<<1, 256, 0, stream>>>(V, sv, h, hval);
    lbo_cvt_kernel<<<(RANK * DIM) / 256, 256, 0, stream>>>(V, Vh, Vt);

    const int nrows  = in_sizes[0] / DIM;       // 4*8192 = 32768
    const int blocks = nrows / ROWS;            // 2048
    lbo_main_kernel<<<blocks, 256, 0, stream>>>(x, scale, bias, alpha, hval, Vh, Vt, out);
}
